// StaircaseSeqAttention_25391846654001
// MI455X (gfx1250) — matrix operand order repaired and checked by reading, not compiled, not run on hardware
//
#include <hip/hip_runtime.h>

typedef __attribute__((ext_vector_type(16))) __bf16 v16bf;
typedef __attribute__((ext_vector_type(8)))  __bf16 v8bf;
typedef __attribute__((ext_vector_type(8)))  float  v8f;

#define BHN   16
#define SEQ   2048
#define DH    64
#define BM    128      // query rows per block (8 waves x 16)
#define BN    32       // keys per iteration
#define NW    8        // waves per block
#define CBUF  256      // circular PE-column buffer (power of 2, > window width 159)
#define SCL   0.1803368801111f   // 0.125 * log2(e): softmax tracked in base-2 domain

__device__ __forceinline__ v8f wmma_bf16(v16bf a, v16bf b, v8f c) {
  return __builtin_amdgcn_wmma_f32_16x16x32_bf16(false, a, false, b, (short)0, c,
                                                 false, false);
}

__global__ __launch_bounds__(256) void staircase_attn_kernel(
    const float* __restrict__ Q, const float* __restrict__ K,
    const float* __restrict__ V, const float* __restrict__ PE,
    float* __restrict__ O) {
  __shared__ __attribute__((aligned(32))) __bf16 sK  [BN * DH];       // 4 KB row-major keys
  __shared__ __attribute__((aligned(32))) __bf16 sVt [DH * BN];       // 4 KB transposed values (perm-K)
  __shared__ __attribute__((aligned(32))) __bf16 sPtc[CBUF * DH];     // 32 KB circular PE (c-major)
  __shared__ __attribute__((aligned(32))) __bf16 sP  [NW * 16 * BN];  // 8 KB per-wave probs (perm-K)

  const int tid    = threadIdx.x;
  const int w      = tid >> 5;
  const int t      = tid & 31;
  const int lane16 = t & 15;
  const int hh     = t >> 4;
  const int bh     = blockIdx.y;
  const int m0     = blockIdx.x * BM;
  const int mw0    = m0 + w * 16;
  const int cb0    = (SEQ - BM) - m0;   // PE window base at n0 = 0 (may be negative)

  // ---- Q tile (16x64) -> two bf16 A operands (contiguous K: lane half h holds k = 16h..16h+15 of each 32-chunk) ----
  const float* qrow = Q + ((size_t)bh * SEQ + (mw0 + lane16)) * DH;
  const int qb = hh * 8;
  v16bf qa[2];
#pragma unroll
  for (int kc = 0; kc < 2; ++kc) {
    union { v16bf v; __bf16 e[16]; } A;
#pragma unroll
    for (int j = 0; j < 8; ++j) {
      A.e[j]     = (__bf16)qrow[kc * 32 + hh * 16 + j];
      A.e[8 + j] = (__bf16)qrow[kc * 32 + hh * 16 + 8 + j];
    }
    qa[kc] = A.v;
  }

  // constant all-ones B operand: P @ ones == per-row sums (permutation-invariant)
  v16bf onesb;
  {
    union { v16bf v; __bf16 e[16]; } U;
#pragma unroll
    for (int j = 0; j < 16; ++j) U.e[j] = (__bf16)1.0f;
    onesb = U.v;
  }

  float mrun[8], lrun[8], alpha[8];
  v8f acc[4];
  const v8f vzero = {0.f, 0.f, 0.f, 0.f, 0.f, 0.f, 0.f, 0.f};
#pragma unroll
  for (int vv = 0; vv < 8; ++vv) { mrun[vv] = -1e30f; lrun[vv] = 0.f; }
#pragma unroll
  for (int dt = 0; dt < 4; ++dt) acc[dt] = vzero;

  // ---- PE prologue: stage columns [cb0, cb0+127] into the circular buffer ----
  {
    const int d  = tid >> 2;           // 0..63
    const int cq = tid & 3;
    const float* per = PE + (size_t)d * SEQ;
#pragma unroll
    for (int kk = 0; kk < 4; ++kk) {
      const int c0 = cb0 + cq * 8 + kk * 32;
      float tmp[8];
#pragma unroll
      for (int j = 0; j < 8; ++j) {
        int c = c0 + j;
        c = c < 0 ? 0 : (c > SEQ - 1 ? SEQ - 1 : c);  // OOB cols are masked-only
        tmp[j] = per[c];
      }
#pragma unroll
      for (int j = 0; j < 8; ++j)
        sPtc[((c0 + j) & (CBUF - 1)) * DH + d] = (__bf16)tmp[j];
    }
  }

  const int n_end = m0 + BM;
  for (int n0 = 0; n0 < n_end; n0 += BN) {
    __syncthreads();                   // previous tiles fully consumed

    // ---- stage K (row-major) and V (transposed with K-permutation) ----
    {
      const int r  = tid >> 3;
      const int cg = (tid & 7) * 8;
      const int rp = ((r & 15) << 1) | (r >> 4);   // perm(k): pairs (c, c+16) adjacent
      const float* kp = K + ((size_t)bh * SEQ + (n0 + r)) * DH + cg;
      const float* vp = V + ((size_t)bh * SEQ + (n0 + r)) * DH + cg;
      const float4 k0 = *(const float4*)kp;
      const float4 k1 = *(const float4*)(kp + 4);
      const float4 v0 = *(const float4*)vp;
      const float4 v1 = *(const float4*)(vp + 4);
      union { v8bf v; __bf16 e[8]; } pk;
      pk.e[0] = (__bf16)k0.x; pk.e[1] = (__bf16)k0.y;
      pk.e[2] = (__bf16)k0.z; pk.e[3] = (__bf16)k0.w;
      pk.e[4] = (__bf16)k1.x; pk.e[5] = (__bf16)k1.y;
      pk.e[6] = (__bf16)k1.z; pk.e[7] = (__bf16)k1.w;
      *(v8bf*)&sK[r * DH + cg] = pk.v;
      sVt[(cg + 0) * BN + rp] = (__bf16)v0.x;
      sVt[(cg + 1) * BN + rp] = (__bf16)v0.y;
      sVt[(cg + 2) * BN + rp] = (__bf16)v0.z;
      sVt[(cg + 3) * BN + rp] = (__bf16)v0.w;
      sVt[(cg + 4) * BN + rp] = (__bf16)v1.x;
      sVt[(cg + 5) * BN + rp] = (__bf16)v1.y;
      sVt[(cg + 6) * BN + rp] = (__bf16)v1.z;
      sVt[(cg + 7) * BN + rp] = (__bf16)v1.w;
      if (n0 + BN < n_end) {           // prefetch next tile into cache
        __builtin_prefetch(kp + (size_t)BN * DH, 0, 1);
        __builtin_prefetch(vp + (size_t)BN * DH, 0, 1);
      }
    }
    // ---- stage the 32 NEW PE columns [cb+128, cb+159] (window slides by BN) ----
    {
      const int cb = cb0 + n0;
      const int d  = tid >> 2;
      const int cq = tid & 3;
      const float* per = PE + (size_t)d * SEQ;
      const int c0 = cb + 128 + cq * 8;
      float tmp[8];
#pragma unroll
      for (int j = 0; j < 8; ++j) {
        int c = c0 + j;
        c = c < 0 ? 0 : (c > SEQ - 1 ? SEQ - 1 : c);
        tmp[j] = per[c];
      }
#pragma unroll
      for (int j = 0; j < 8; ++j)
        sPtc[((c0 + j) & (CBUF - 1)) * DH + d] = (__bf16)tmp[j];
    }
    __syncthreads();

    if (n0 <= mw0 + 15) {              // wave-uniform causal tile filter
      // ---- S = Q @ K^T ----
      v8f s[2]; s[0] = vzero; s[1] = vzero;
#pragma unroll
      for (int ns = 0; ns < 2; ++ns)
#pragma unroll
        for (int kc = 0; kc < 2; ++kc) {
          const v16bf b = *(const v16bf*)&sK[(ns * 16 + lane16) * DH + kc * 32 + hh * 16];
          s[ns] = wmma_bf16(qa[kc], b, s[ns]);
        }

      // ---- R = Q @ PE_window : 16x48, B rows from circular buffer ----
      const int cwb = cb0 + n0 + 112 - 16 * w;   // wave's logical window base
      v8f r[3];
#pragma unroll
      for (int cs = 0; cs < 3; ++cs) {
        r[cs] = vzero;
        const int prow = ((cwb + cs * 16 + lane16) & (CBUF - 1)) * DH;
#pragma unroll
        for (int kc = 0; kc < 2; ++kc) {
          const v16bf b = *(const v16bf*)&sPtc[prow + kc * 32 + hh * 16];
          r[cs] = wmma_bf16(qa[kc], b, r[cs]);
        }
      }

      // ---- skew gather via cross-lane shuffles; mask only on diagonal tiles ----
      if (n0 + BN - 1 <= mw0) {        // interior tile: fully unmasked
#pragma unroll
        for (int vv = 0; vv < 8; ++vv) {
          const int u    = lane16 + 15 - vv - 8 * hh;     // in [0,30]
          const int srcL = 16 * hh + (u & 15);
          const float t0 = __shfl(r[0][vv], srcL, 32);
          const float t1 = __shfl(r[1][vv], srcL, 32);
          const float t2 = __shfl(r[2][vv], srcL, 32);
          const float b0 = (u < 16) ? t0 : t1;
          const float b1 = (u < 16) ? t1 : t2;
          s[0][vv] = (s[0][vv] + b0) * SCL;
          s[1][vv] = (s[1][vv] + b1) * SCL;
        }
      } else {                         // diagonal tile: apply causal mask
#pragma unroll
        for (int vv = 0; vv < 8; ++vv) {
          const int i    = mw0 + vv + 8 * hh;
          const int u    = lane16 + 15 - vv - 8 * hh;
          const int srcL = 16 * hh + (u & 15);
          const float t0 = __shfl(r[0][vv], srcL, 32);
          const float t1 = __shfl(r[1][vv], srcL, 32);
          const float t2 = __shfl(r[2][vv], srcL, 32);
          const float b0 = (u < 16) ? t0 : t1;
          const float b1 = (u < 16) ? t1 : t2;
          const int l0 = n0 + lane16;
          s[0][vv] = (l0      <= i) ? (s[0][vv] + b0) * SCL : -1e30f;
          s[1][vv] = (l0 + 16 <= i) ? (s[1][vv] + b1) * SCL : -1e30f;
        }
      }

      // ---- online softmax in base-2 domain: raw v_exp_f32, no range fixup ----
#pragma unroll
      for (int vv = 0; vv < 8; ++vv) {
        float pm = fmaxf(s[0][vv], s[1][vv]);
#pragma unroll
        for (int off = 1; off < 16; off <<= 1)
          pm = fmaxf(pm, __shfl_xor(pm, off, 16));
        const float mn = fmaxf(mrun[vv], pm);
        const float al = __builtin_amdgcn_exp2f(mrun[vv] - mn);
        s[0][vv] = __builtin_amdgcn_exp2f(s[0][vv] - mn);
        s[1][vv] = __builtin_amdgcn_exp2f(s[1][vv] - mn);
        mrun[vv]  = mn;
        alpha[vv] = al;
#pragma unroll
        for (int dt = 0; dt < 4; ++dt) acc[dt][vv] *= al;
      }

      // ---- C-layout -> A-layout via per-wave LDS, packed b32 stores (perm-K) ----
      __bf16* Pw = &sP[w * 16 * BN];
#pragma unroll
      for (int vv = 0; vv < 8; ++vv) {
        union { __bf16 e[2]; unsigned int u; } pr;
        pr.e[0] = (__bf16)s[0][vv];    // perm pos 2*lane16
        pr.e[1] = (__bf16)s[1][vv];    // perm pos 2*lane16+1
        *(unsigned int*)&Pw[(vv + 8 * hh) * BN + lane16 * 2] = pr.u;
      }

      union { v16bf v; v8bf h2[2]; } Ap;
      Ap.h2[0] = *(const v8bf*)&Pw[lane16 * BN + hh * 16];
      Ap.h2[1] = *(const v8bf*)&Pw[lane16 * BN + hh * 16 + 8];

      // ---- row sums via WMMA with all-ones B ----
      const v8f rs = wmma_bf16(Ap.v, onesb, vzero);

      // ---- acc += P @ V (same perm-K on both operands) ----
#pragma unroll
      for (int dt = 0; dt < 4; ++dt) {
        const v16bf b = *(const v16bf*)&sVt[(dt * 16 + lane16) * BN + hh * 16];
        acc[dt] = wmma_bf16(Ap.v, b, acc[dt]);
      }
#pragma unroll
      for (int vv = 0; vv < 8; ++vv)
        lrun[vv] = lrun[vv] * alpha[vv] + rs[vv];
    }
  }

  // ---- epilogue: out = acc / l ----
#pragma unroll
  for (int vv = 0; vv < 8; ++vv) {
    const float inv = __builtin_amdgcn_rcpf(lrun[vv]);
    const int i = mw0 + vv + 8 * hh;
    float* orow = O + ((size_t)bh * SEQ + i) * DH;
#pragma unroll
    for (int dt = 0; dt < 4; ++dt)
      orow[dt * 16 + lane16] = acc[dt][vv] * inv;
  }
}

extern "C" void kernel_launch(void* const* d_in, const int* in_sizes, int n_in,
                              void* d_out, int out_size, void* d_ws, size_t ws_size,
                              hipStream_t stream) {
  (void)in_sizes; (void)n_in; (void)out_size; (void)d_ws; (void)ws_size;
  const float* q  = (const float*)d_in[0];
  const float* k  = (const float*)d_in[1];
  const float* v  = (const float*)d_in[2];
  const float* pe = (const float*)d_in[3];
  float* out = (float*)d_out;
  dim3 grid(SEQ / BM, BHN);
  staircase_attn_kernel<<<grid, 256, 0, stream>>>(q, k, v, pe, out);
}
